// KimiK25TextTensorizedMoE_55662776156365
// MI455X (gfx1250) — matrix operand order repaired and checked by reading, not compiled, not run on hardware
//
#include <hip/hip_runtime.h>
#include <stdint.h>

// Problem constants (match reference)
#define TT    2048        // B*S tokens
#define HH    1024        // hidden
#define II    512         // expert intermediate
#define EE    32          // experts
#define TOPK  4
#define NGRP  4
#define GSZ   8           // E / NGRP
#define SH_II 1024        // shared intermediate
#define SCALE_F 2.5f
#define MT    32          // token tile (two 16-row WMMA sub-tiles)

typedef __attribute__((ext_vector_type(16))) _Float16 v16h;
typedef __attribute__((ext_vector_type(8)))  _Float16 v8h;
typedef __attribute__((ext_vector_type(8)))  float    v8f;
typedef __attribute__((ext_vector_type(4)))  float    v4f;
typedef __attribute__((ext_vector_type(4)))  unsigned int v4u;
typedef __attribute__((ext_vector_type(8)))  int      v8i;
typedef __attribute__((ext_vector_type(4)))  int      v4i;

// ---------------------------------------------------------------------------
// WMMA helper
// ---------------------------------------------------------------------------
__device__ __forceinline__ v8f wmma_f16(v16h a, v16h b, v8f c) {
  return __builtin_amdgcn_wmma_f32_16x16x32_f16(
      false, a, false, b, (short)0, c, false, false);
}

// A fragment (16x32 f16) from LDS row-major [16][ldk] tile at column k0.
__device__ __forceinline__ v16h load_a_lds(const _Float16* A, int ldk, int k0, int lane) {
  int m  = lane & 15;
  int kb = (lane >> 4) << 4;             // 0 or 16
  const _Float16* p = A + m * ldk + k0 + kb;
  v8h lo = *(const v8h*)(p);             // ds_load_b128
  v8h hi = *(const v8h*)(p + 8);         // ds_load_b128
  v16h a;
#pragma unroll
  for (int i = 0; i < 8; ++i) { a[i] = lo[i]; a[8 + i] = hi[i]; }
  return a;
}

// B fragment (32x16 f16): B[k][n] = W[n0+n][k], W row-major [N][ldk] f32 global.
// Each lane: 64 contiguous bytes (4x global_load_b128) + cvt to f16.
__device__ __forceinline__ v16h load_b_glb(const float* __restrict__ W, int ldk,
                                           int n0, int k0, int lane) {
  int n  = lane & 15;
  int kb = (lane >> 4) << 4;             // 0 or 16
  const float* p = W + (size_t)(n0 + n) * ldk + (k0 + kb);
  v16h b;
#pragma unroll
  for (int q = 0; q < 4; ++q) {
    v4f f = *(const v4f*)(p + q * 4);
#pragma unroll
    for (int i = 0; i < 4; ++i) b[q * 4 + i] = (_Float16)f[i];
  }
  return b;
}

__device__ __forceinline__ float silu_f(float g) {
  return g / (1.f + __expf(-g));
}

// ---------------------------------------------------------------------------
// TDM descriptors (ISA 8.3-8.7).  lds byte offset = low 32 bits of generic VA.
// Toolchain here exposes the 6-arg builtin:
//   (uint32x4 g0, int32x8 g1, int32x4 g2, int32x4 g3, int32x8, i32 cpol)
// 5th operand semantics uncharacterized -> zero-filled (probe-verified form).
// ---------------------------------------------------------------------------
__device__ __forceinline__ v8i v8i_zero() {
  v8i z;
#pragma unroll
  for (int i = 0; i < 8; ++i) z[i] = 0;
  return z;
}

// Gather mode: up to 16 rows of a [TT][HH] f16 tensor by 16-bit row index.
__device__ __forceinline__ void tdm_gather16(unsigned lds_off,
                                             const _Float16* gbase,
                                             const int* idx, int nrows) {
  unsigned long long ga = (unsigned long long)(uintptr_t)gbase;
  v4u g0;
  g0[0] = 0x80000001u;                         // count=1, gather_mode, 16b idx
  g0[1] = lds_off;                             // D#.lds_addr
  g0[2] = (unsigned)(ga & 0xffffffffu);        // global_addr[31:0]
  g0[3] = (unsigned)((ga >> 32) & 0x01ffffffu) | (2u << 30);   // [56:32]|type=2
  v8i g1;
  g1[0] = (1 << 16);                           // data_size = 1 -> 2 bytes
  g1[1] = (int)((unsigned)HH << 16);           // tensor_dim0 = 1024 (lo16)
  g1[2] = (int)((unsigned)TT << 16);           // tensor_dim1 = 2048 (lo16)
  g1[3] = (int)((unsigned)HH << 16);           // tile_dim0 = 1024
  g1[4] = nrows;                               // tile_dim1 = #valid indices
  g1[5] = HH;                                  // tensor_dim0_stride = 1024
  g1[6] = 0;
  g1[7] = 0;
  v4i g2, g3;
#pragma unroll
  for (int p = 0; p < 4; ++p) {
    g2[p] = (idx[2 * p] & 0xffff)     | (idx[2 * p + 1] << 16);
    g3[p] = (idx[8 + 2 * p] & 0xffff) | (idx[8 + 2 * p + 1] << 16);
  }
  __builtin_amdgcn_tensor_load_to_lds(g0, g1, g2, g3, v8i_zero(), 0);
}

// Plain 2-D tile load: nrows contiguous rows of [rows][HH] f16 starting @grow0.
__device__ __forceinline__ void tdm_load2d(unsigned lds_off,
                                           const _Float16* grow0, int nrows) {
  unsigned long long ga = (unsigned long long)(uintptr_t)grow0;
  v4u g0;
  g0[0] = 1u;                                  // count=1
  g0[1] = lds_off;
  g0[2] = (unsigned)(ga & 0xffffffffu);
  g0[3] = (unsigned)((ga >> 32) & 0x01ffffffu) | (2u << 30);
  v8i g1;
  g1[0] = (1 << 16);                           // data_size = 2 bytes
  g1[1] = (int)((unsigned)HH << 16);           // tensor_dim0 = 1024
  g1[2] = (int)((unsigned)nrows << 16);        // tensor_dim1 = nrows
  g1[3] = (int)((unsigned)HH << 16);           // tile_dim0 = 1024
  g1[4] = nrows;                               // tile_dim1
  g1[5] = HH;                                  // tensor_dim0_stride
  g1[6] = 0;
  g1[7] = 0;
  v4i z4; z4[0] = 0; z4[1] = 0; z4[2] = 0; z4[3] = 0;
  __builtin_amdgcn_tensor_load_to_lds(g0, g1, z4, z4, v8i_zero(), 0);
}

// ---------------------------------------------------------------------------
__global__ void zero_cnt_kernel(int* __restrict__ cnt) {
  if (threadIdx.x < EE) cnt[threadIdx.x] = 0;
}

__global__ void cvt_f16_kernel(const float* __restrict__ in,
                               _Float16* __restrict__ out, int n) {
  int i = blockIdx.x * blockDim.x + threadIdx.x;
  if (i < n) out[i] = (_Float16)in[i];
}

// ---------------------------------------------------------------------------
// Router + scatter: one wave32 per token, lane == expert (E == 32).
// ---------------------------------------------------------------------------
__global__ void router_scatter_kernel(const float* __restrict__ tokens,
                                      const float* __restrict__ rw,
                                      const float* __restrict__ rb,
                                      int*  __restrict__ cnt,
                                      int*  __restrict__ list,
                                      float* __restrict__ wlist) {
  __shared__ float s_sfc[8][EE];
  __shared__ float s_sc[8][EE];
  const int wv   = threadIdx.x >> 5;
  const int lane = threadIdx.x & 31;
  const int tok  = blockIdx.x * 8 + wv;

  const float4* x = (const float4*)(tokens + (size_t)tok * HH);
  const float4* w = (const float4*)(rw + (size_t)lane * HH);
  float acc = 0.f;
  for (int i = 0; i < HH / 4; ++i) {
    float4 a = x[i], b = w[i];
    acc += a.x * b.x + a.y * b.y + a.z * b.z + a.w * b.w;
  }
  float score = 1.f / (1.f + __expf(-acc));
  s_sc[wv][lane]  = score;
  s_sfc[wv][lane] = score + rb[lane];
  __syncthreads();
  if (lane != 0) return;

  float gs[NGRP];
  for (int g = 0; g < NGRP; ++g) {
    float m1 = -1e30f; int i1 = -1;
    for (int i = 0; i < GSZ; ++i) {
      float v = s_sfc[wv][g * GSZ + i];
      if (v > m1) { m1 = v; i1 = i; }
    }
    float m2 = -1e30f;
    for (int i = 0; i < GSZ; ++i) {
      if (i == i1) continue;
      float v = s_sfc[wv][g * GSZ + i];
      if (v > m2) m2 = v;
    }
    gs[g] = m1 + m2;
  }
  int g1 = 0; float b1 = gs[0];
  for (int g = 1; g < NGRP; ++g) if (gs[g] > b1) { b1 = gs[g]; g1 = g; }
  int g2 = -1; float b2 = -1e30f;
  for (int g = 0; g < NGRP; ++g) {
    if (g == g1) continue;
    if (gs[g] > b2) { b2 = gs[g]; g2 = g; }
  }
  int sel[TOPK]; float wsel[TOPK]; float wsum = 0.f;
  unsigned chosen = 0u;
  for (int k = 0; k < TOPK; ++k) {
    float bv = -1e30f; int bi = 0;
    for (int e = 0; e < EE; ++e) {
      int g = e >> 3;
      if (g != g1 && g != g2) continue;
      if (chosen & (1u << e)) continue;
      float v = s_sfc[wv][e];
      if (v > bv) { bv = v; bi = e; }
    }
    chosen |= (1u << bi);
    sel[k]  = bi;
    wsel[k] = s_sc[wv][bi];
    wsum   += wsel[k];
  }
  float inv = SCALE_F / (wsum + 1e-20f);
  for (int k = 0; k < TOPK; ++k) {
    int e = sel[k];
    int pos = atomicAdd(&cnt[e], 1);
    list[e * TT + pos]  = (tok << 2) | k;
    wlist[e * TT + pos] = wsel[k] * inv;
  }
}

// ---------------------------------------------------------------------------
// Per-expert fused SwiGLU GEMM on a 32-token tile.  grid = (E, T/32).
// A tile staged by TDM gather (16 rows per descriptor, token ids as indices).
// ---------------------------------------------------------------------------
__global__ void expert_kernel(const _Float16* __restrict__ Xh,
                              const float* __restrict__ gate_w,
                              const float* __restrict__ up_w,
                              const float* __restrict__ down_w,
                              const int*  __restrict__ cnt,
                              const int*  __restrict__ list,
                              const float* __restrict__ wlist,
                              float* __restrict__ scratch) {
  const int e    = blockIdx.x;
  const int tile = blockIdx.y;
  const int Ce   = cnt[e];
  if (tile * MT >= Ce) return;

  __shared__ __attribute__((aligned(128))) _Float16 lA[MT * HH]; // 64 KB
  __shared__ __attribute__((aligned(128))) _Float16 lH[MT * II]; // 32 KB
  __shared__ int   s_tok4[MT];
  __shared__ float s_wt[MT];

  const int t = threadIdx.x;
  const int base = e * TT + tile * MT;
  if (t < MT) {
    int r = tile * MT + t;
    if (r < Ce) {
      s_tok4[t] = list[base + t];
      s_wt[t]   = wlist[base + t];
    } else {
      s_tok4[t] = -1;
      s_wt[t]   = 0.f;
    }
  }
  const int wv = t >> 5, lane = t & 31;
  if (wv == 0) {
    // stage A via TDM gather: two 16-row descriptors
    int navail = Ce - tile * MT;
    int vcnt0 = navail < 16 ? navail : 16;
    int rem = navail - 16;
    int vcnt1 = rem < 0 ? 0 : (rem > 16 ? 16 : rem);
    unsigned lA_off = (unsigned)(uintptr_t)(void*)lA;  // LDS byte offset
    int idx[16];
    for (int r = 0; r < 16; ++r)
      idx[r] = (r < navail) ? (list[base + r] >> 2) : 0;
    tdm_gather16(lA_off, Xh, idx, vcnt0);
    if (vcnt1 > 0) {
      for (int r = 0; r < 16; ++r)
        idx[r] = (16 + r < navail) ? (list[base + 16 + r] >> 2) : 0;
      tdm_gather16(lA_off + 16 * HH * 2, Xh, idx, vcnt1);
    }
    __builtin_amdgcn_s_wait_tensorcnt(0);
  }
  __syncthreads();

  const float* gw = gate_w + (size_t)e * II * HH;
  const float* uw = up_w   + (size_t)e * II * HH;
  const int mBase = (lane >> 4) << 3;
  const int nloc  = lane & 15;

  // Phase 1: gate/up [32 x 512]; per wave: 4 column tiles in 2 passes.
  for (int g = 0; g < 2; ++g) {
    v8f vg0[2] = {{}, {}}, vg1[2] = {{}, {}};
    v8f vu0[2] = {{}, {}}, vu1[2] = {{}, {}};
    for (int k0 = 0; k0 < HH; k0 += 32) {
      v16h a0 = load_a_lds(lA, HH, k0, lane);
      v16h a1 = load_a_lds(lA + 16 * HH, HH, k0, lane);
#pragma unroll
      for (int j = 0; j < 2; ++j) {
        int n0 = (wv * 4 + g * 2 + j) * 16;
        v16h bg = load_b_glb(gw, HH, n0, k0, lane);
        vg0[j] = wmma_f16(a0, bg, vg0[j]);
        vg1[j] = wmma_f16(a1, bg, vg1[j]);
        v16h bu = load_b_glb(uw, HH, n0, k0, lane);
        vu0[j] = wmma_f16(a0, bu, vu0[j]);
        vu1[j] = wmma_f16(a1, bu, vu1[j]);
      }
    }
#pragma unroll
    for (int j = 0; j < 2; ++j) {
      int ncol = (wv * 4 + g * 2 + j) * 16 + nloc;
#pragma unroll
      for (int r = 0; r < 8; ++r) {
        lH[(mBase + r) * II + ncol]      = (_Float16)(silu_f(vg0[j][r]) * vu0[j][r]);
        lH[(16 + mBase + r) * II + ncol] = (_Float16)(silu_f(vg1[j][r]) * vu1[j][r]);
      }
    }
  }
  __syncthreads();

  // Phase 2: down projection [32 x 1024]; per wave: 8 column tiles.
  const float* dw = down_w + (size_t)e * HH * II;
  for (int j = 0; j < 8; ++j) {
    int h0 = (wv * 8 + j) * 16;
    v8f acc0 = {}, acc1 = {};
    for (int k0 = 0; k0 < II; k0 += 32) {
      v16h a0 = load_a_lds(lH, II, k0, lane);
      v16h a1 = load_a_lds(lH + 16 * II, II, k0, lane);
      v16h b  = load_b_glb(dw, II, h0, k0, lane);
      acc0 = wmma_f16(a0, b, acc0);
      acc1 = wmma_f16(a1, b, acc1);
    }
#pragma unroll
    for (int r = 0; r < 8; ++r) {
      int m0 = mBase + r, m1 = 16 + mBase + r;
      int tk0 = s_tok4[m0], tk1 = s_tok4[m1];
      if (tk0 >= 0) scratch[(size_t)tk0 * HH + h0 + nloc] = acc0[r] * s_wt[m0];
      if (tk1 >= 0) scratch[(size_t)tk1 * HH + h0 + nloc] = acc1[r] * s_wt[m1];
    }
  }
}

// ---------------------------------------------------------------------------
// Shared experts: dense fused SwiGLU over all tokens; writes d_out.
// A tile staged by plain 2-D TDM load.  grid = T/32.
// ---------------------------------------------------------------------------
__global__ void shared_expert_kernel(const _Float16* __restrict__ Xh,
                                     const float* __restrict__ sgw,
                                     const float* __restrict__ suw,
                                     const float* __restrict__ sdw,
                                     float* __restrict__ out) {
  const int tile = blockIdx.x;                             // T/32 tiles
  __shared__ __attribute__((aligned(128))) _Float16 lA[MT * HH];    // 64 KB
  __shared__ __attribute__((aligned(128))) _Float16 lH[MT * SH_II]; // 64 KB

  const int t = threadIdx.x;
  const int wv = t >> 5, lane = t & 31;
  if (wv == 0) {
    unsigned lA_off = (unsigned)(uintptr_t)(void*)lA;
    tdm_load2d(lA_off, Xh + (size_t)tile * MT * HH, MT);
    __builtin_amdgcn_s_wait_tensorcnt(0);
  }
  __syncthreads();

  const int mBase = (lane >> 4) << 3;
  const int nloc  = lane & 15;

  // Phase 1: gate/up [32 x 1024]; per wave: 8 column tiles in 4 passes.
  for (int g = 0; g < 4; ++g) {
    v8f vg0[2] = {{}, {}}, vg1[2] = {{}, {}};
    v8f vu0[2] = {{}, {}}, vu1[2] = {{}, {}};
    for (int k0 = 0; k0 < HH; k0 += 32) {
      v16h a0 = load_a_lds(lA, HH, k0, lane);
      v16h a1 = load_a_lds(lA + 16 * HH, HH, k0, lane);
#pragma unroll
      for (int j = 0; j < 2; ++j) {
        int n0 = (wv * 8 + g * 2 + j) * 16;
        v16h bg = load_b_glb(sgw, HH, n0, k0, lane);
        vg0[j] = wmma_f16(a0, bg, vg0[j]);
        vg1[j] = wmma_f16(a1, bg, vg1[j]);
        v16h bu = load_b_glb(suw, HH, n0, k0, lane);
        vu0[j] = wmma_f16(a0, bu, vu0[j]);
        vu1[j] = wmma_f16(a1, bu, vu1[j]);
      }
    }
#pragma unroll
    for (int j = 0; j < 2; ++j) {
      int ncol = (wv * 8 + g * 2 + j) * 16 + nloc;
#pragma unroll
      for (int r = 0; r < 8; ++r) {
        lH[(mBase + r) * SH_II + ncol]      = (_Float16)(silu_f(vg0[j][r]) * vu0[j][r]);
        lH[(16 + mBase + r) * SH_II + ncol] = (_Float16)(silu_f(vg1[j][r]) * vu1[j][r]);
      }
    }
  }
  __syncthreads();

  // Phase 2: down projection [32 x 1024]; per wave: 8 column tiles.
  for (int j = 0; j < 8; ++j) {
    int h0 = (wv * 8 + j) * 16;
    v8f acc0 = {}, acc1 = {};
    for (int k0 = 0; k0 < SH_II; k0 += 32) {
      v16h a0 = load_a_lds(lH, SH_II, k0, lane);
      v16h a1 = load_a_lds(lH + 16 * SH_II, SH_II, k0, lane);
      v16h b  = load_b_glb(sdw, SH_II, h0, k0, lane);
      acc0 = wmma_f16(a0, b, acc0);
      acc1 = wmma_f16(a1, b, acc1);
    }
#pragma unroll
    for (int r = 0; r < 8; ++r) {
      int m0 = tile * MT + mBase + r;
      int m1 = tile * MT + 16 + mBase + r;
      out[(size_t)m0 * HH + h0 + nloc] = acc0[r];
      out[(size_t)m1 * HH + h0 + nloc] = acc1[r];
    }
  }
}

// ---------------------------------------------------------------------------
// Combine: out += sum_k routed contribution (fixed order -> deterministic)
// ---------------------------------------------------------------------------
__global__ void combine_kernel(const float* __restrict__ scratch,
                               float* __restrict__ out, int n4) {
  int i = blockIdx.x * blockDim.x + threadIdx.x;
  if (i >= n4) return;
  int tok = i >> 8;                       // HH/4 float4 per row
  int c   = i & 255;
  size_t base = (size_t)tok * 4 * (HH / 4) + c;
  const float4* s4 = (const float4*)scratch;
  float4 v = ((const float4*)out)[i];
#pragma unroll
  for (int k = 0; k < 4; ++k) {
    float4 s = s4[base + (size_t)k * (HH / 4)];
    v.x += s.x; v.y += s.y; v.z += s.z; v.w += s.w;
  }
  ((float4*)out)[i] = v;
}

// ---------------------------------------------------------------------------
extern "C" void kernel_launch(void* const* d_in, const int* in_sizes, int n_in,
                              void* d_out, int out_size, void* d_ws, size_t ws_size,
                              hipStream_t stream) {
  (void)in_sizes; (void)n_in; (void)out_size; (void)ws_size;
  const float* hidden = (const float*)d_in[0];
  const float* rw     = (const float*)d_in[1];
  const float* rb     = (const float*)d_in[2];
  const float* gw     = (const float*)d_in[3];
  const float* uw     = (const float*)d_in[4];
  const float* dw     = (const float*)d_in[5];
  const float* sgw    = (const float*)d_in[6];
  const float* suw    = (const float*)d_in[7];
  const float* sdw    = (const float*)d_in[8];
  float* out = (float*)d_out;

  // ws: cnt[E] | list[E*T] | wlist[E*T] | Xh[T*H] f16 | scratch[T*4*H] f32
  char* ws = (char*)d_ws;
  int*      cnt     = (int*)(ws);
  int*      list    = (int*)(ws + 256);
  float*    wlist   = (float*)(ws + 256 + (size_t)EE * TT * 4);
  _Float16* Xh      = (_Float16*)(ws + 256 + 2 * (size_t)EE * TT * 4);
  float*    scratch = (float*)(ws + 256 + 2 * (size_t)EE * TT * 4 +
                               (size_t)TT * HH * 2);

  zero_cnt_kernel<<<1, 32, 0, stream>>>(cnt);
  cvt_f16_kernel<<<(TT * HH + 255) / 256, 256, 0, stream>>>(hidden, Xh, TT * HH);
  router_scatter_kernel<<<TT / 8, 256, 0, stream>>>(hidden, rw, rb, cnt, list, wlist);
  shared_expert_kernel<<<TT / MT, 256, 0, stream>>>(Xh, sgw, suw, sdw, out);
  expert_kernel<<<dim3(EE, TT / MT), 256, 0, stream>>>(Xh, gw, uw, dw, cnt, list,
                                                       wlist, scratch);
  combine_kernel<<<(TT * HH / 4 + 255) / 256, 256, 0, stream>>>(scratch, out,
                                                                TT * HH / 4);
}
